// ExpertVectorSystem_68831145886300
// MI455X (gfx1250) — matrix operand order repaired and checked by reading, not compiled, not run on hardware
//
#include <hip/hip_runtime.h>
#include <math.h>

typedef __attribute__((ext_vector_type(16))) __bf16 v16bf;
typedef __attribute__((ext_vector_type(8)))  float  v8f;

#define BATCH 16384
#define OUTCOLS 5632

static __device__ __forceinline__ unsigned short f32_to_bf16(float f) {
  union { float f; unsigned int u; } v; v.f = f;
  unsigned int u = v.u;
  u += 0x7FFFu + ((u >> 16) & 1u);   // round-to-nearest-even
  return (unsigned short)(u >> 16);
}

static __device__ __forceinline__ float fast_tanh(float x) {
#if __has_builtin(__builtin_amdgcn_tanhf)
  return __builtin_amdgcn_tanhf(x);        // native v_tanh_f32 on gfx1250
#elif __has_builtin(__builtin_amdgcn_tanh_f32)
  return __builtin_amdgcn_tanh_f32(x);
#else
  return tanhf(x);
#endif
}

union FragB { v16bf v; uint4 q[2]; };

struct Params {
  const unsigned short* we;    // [16384 x 64] bf16
  const unsigned short* w1t;   // per-layer [2r x 64] bf16, packed
  const unsigned short* w2t;   // per-layer [r x 2r] bf16, packed
  const float* b1[8];
  const float* b2[8];
  float* out;                  // [16384 x 5632] f32
};

__device__ static constexpr int c_rank[8]   = {256,384,512,640,768,896,1024,1152};
__device__ static constexpr int c_w1off[8]  = {0,32768,81920,147456,229376,327680,442368,573440};
__device__ static constexpr int c_w2off[8]  = {0,131072,425984,950272,1769472,2949120,4554752,6651904};
__device__ static constexpr int c_outoff[8] = {0,256,640,1152,1792,2560,3456,4480};

// ---------------- kernel 1: we = expert_weights @ expert_vectors (bf16 out) --
__global__ void we_kernel(const float* __restrict__ ew, const float* __restrict__ ev,
                          unsigned short* __restrict__ we) {
  int idx = blockIdx.x * blockDim.x + threadIdx.x;   // over 16384*64
  int b = idx >> 6, d = idx & 63;
  float s = 0.f;
#pragma unroll
  for (int e = 0; e < 16; ++e) s = fmaf(ew[b * 16 + e], ev[e * 64 + d], s);
  we[idx] = f32_to_bf16(s);
}

// ---------------- kernel 2: tiled transpose + f32->bf16 convert --------------
// in: [rows x cols] f32 row-major;  out: [cols x rows] bf16 row-major
__global__ void transpose_kernel(const float* __restrict__ in, unsigned short* __restrict__ out,
                                 int rows, int cols) {
  __shared__ float tile[32][33];
  int c0 = blockIdx.x * 32, r0 = blockIdx.y * 32;
  int tx = threadIdx.x, ty = threadIdx.y;            // 32 x 8
#pragma unroll
  for (int i = 0; i < 32; i += 8)
    tile[ty + i][tx] = in[(r0 + ty + i) * cols + (c0 + tx)];
  __syncthreads();
#pragma unroll
  for (int i = 0; i < 32; i += 8)
    out[(c0 + ty + i) * rows + (r0 + tx)] = f32_to_bf16(tile[tx][ty + i]);
}

// ---------------- kernel 3: fused  relu(we@w1+b1) @ w2 + b2 -> tanh*0.1 ------
// grid: (BATCH/16, 8 layers), block: 256 threads = 8 waves.
// LDS h: [16 x (2r+8)] bf16, produced in phase 1, consumed in phase 2.
// All loop control is scalar (wave id in SGPR) so EXEC stays all-1s for WMMA.
__launch_bounds__(256)
__global__ void fused_kernel(Params p) {
  extern __shared__ unsigned short h[];
  const int l     = blockIdx.y;
  const int r     = c_rank[l];
  const int r2    = 2 * r;
  const int hp    = r2 + 8;                 // padded pitch (elements)
  const int mbase = blockIdx.x * 16;
  const int lane  = threadIdx.x & 31;
  const int wave  = __builtin_amdgcn_readfirstlane(threadIdx.x >> 5);  // SGPR
  const int n     = lane & 15;              // col (B/C/D) or row (A) within tile
  const int hi    = lane >> 4;              // lane half
  const int koffA = hi * 8;                 // unused; A loads also use koffB
  const int koffB = hi * 16;                // B-matrix K-chunk select

  const unsigned short* __restrict__ w1t = p.w1t + c_w1off[l];  // [2r x 64]
  const unsigned short* __restrict__ w2t = p.w2t + c_w2off[l];  // [r x 2r]
  const float* __restrict__ b1 = p.b1[l];
  const float* __restrict__ b2 = p.b2[l];

  // A fragments for the we tile (row-major [16384 x 64] bf16), K=64 -> 2 k-steps
  FragB a0, a1;
  const unsigned short* werow = p.we + (size_t)(mbase + n) * 64;
  a0.q[0] = *(const uint4*)(werow + 0  + koffB);
  a0.q[1] = *(const uint4*)(werow + 8  + koffB);
  a1.q[0] = *(const uint4*)(werow + 32 + koffB);
  a1.q[1] = *(const uint4*)(werow + 40 + koffB);

  // ---- phase 1: h = relu(we @ w1 + b1), bf16 into LDS ----
  const int nt1 = r2 >> 4;
  for (int j = wave; j < nt1; j += 8) {
    const int jc = j << 4;
    v8f acc = {0.f, 0.f, 0.f, 0.f, 0.f, 0.f, 0.f, 0.f};
    const unsigned short* brow = w1t + (size_t)(jc + n) * 64;   // lane = column
    FragB b0f, b1f;
    b0f.q[0] = *(const uint4*)(brow + 0  + koffB);
    b0f.q[1] = *(const uint4*)(brow + 8  + koffB);
    b1f.q[0] = *(const uint4*)(brow + 32 + koffB);
    b1f.q[1] = *(const uint4*)(brow + 40 + koffB);
    acc = __builtin_amdgcn_wmma_f32_16x16x32_bf16(false, a0.v, false, b0f.v, (short)0, acc, false, false);
    acc = __builtin_amdgcn_wmma_f32_16x16x32_bf16(false, a1.v, false, b1f.v, (short)0, acc, false, false);
    const float bias = b1[jc + n];
    unsigned short* hcol = h + hi * 8 * hp + jc + n;
#pragma unroll
    for (int v = 0; v < 8; ++v) {
      float x = acc[v] + bias;
      hcol[v * hp] = f32_to_bf16(fmaxf(x, 0.0f));
    }
  }
  __syncthreads();

  // ---- phase 2: out = tanh(h @ w2 + b2) * 0.1 ----
  // 2 column tiles per wave-iteration: two independent WMMA accumulation
  // chains (hides D->C hazard) and each LDS A-fragment feeds 2 WMMAs.
  const int nt2 = r >> 4;
  const int colbase = c_outoff[l];
  const unsigned short* arow = h + n * hp;                       // lane = row of A
  for (int j = wave; j < nt2; j += 16) {
    const int  jc0 = j << 4;
    const bool two = (j + 8) < nt2;             // uniform per wave
    const int  jc1 = two ? jc0 + 128 : jc0;
    v8f acc0 = {0.f, 0.f, 0.f, 0.f, 0.f, 0.f, 0.f, 0.f};
    v8f acc1 = {0.f, 0.f, 0.f, 0.f, 0.f, 0.f, 0.f, 0.f};
    const unsigned short* brow0 = w2t + (size_t)(jc0 + n) * r2;  // lane = column
    const unsigned short* brow1 = w2t + (size_t)(jc1 + n) * r2;
#pragma unroll 2
    for (int k0 = 0; k0 < r2; k0 += 32) {
      FragB af, bf0, bf1;
      af.q[0]  = *(const uint4*)(arow + k0 + koffB);      // LDS ds_load_b128
      af.q[1]  = *(const uint4*)(arow + k0 + 8 + koffB);
      bf0.q[0] = *(const uint4*)(brow0 + k0 + koffB);     // global, L2-resident
      bf0.q[1] = *(const uint4*)(brow0 + k0 + 8 + koffB);
      bf1.q[0] = *(const uint4*)(brow1 + k0 + koffB);
      bf1.q[1] = *(const uint4*)(brow1 + k0 + 8 + koffB);
      acc0 = __builtin_amdgcn_wmma_f32_16x16x32_bf16(false, af.v, false, bf0.v, (short)0, acc0, false, false);
      acc1 = __builtin_amdgcn_wmma_f32_16x16x32_bf16(false, af.v, false, bf1.v, (short)0, acc1, false, false);
    }
    {
      const float bias = b2[jc0 + n];
      float* orow = p.out + (size_t)(mbase + hi * 8) * OUTCOLS + colbase + jc0 + n;
#pragma unroll
      for (int v = 0; v < 8; ++v)
        orow[(size_t)v * OUTCOLS] = fast_tanh(acc0[v] + bias) * 0.1f;
    }
    if (two) {
      const float bias = b2[jc1 + n];
      float* orow = p.out + (size_t)(mbase + hi * 8) * OUTCOLS + colbase + jc1 + n;
#pragma unroll
      for (int v = 0; v < 8; ++v)
        orow[(size_t)v * OUTCOLS] = fast_tanh(acc1[v] + bias) * 0.1f;
    }
  }
}

extern "C" void kernel_launch(void* const* d_in, const int* in_sizes, int n_in,
                              void* d_out, int out_size, void* d_ws, size_t ws_size,
                              hipStream_t stream) {
  static const int ranks[8] = {256, 384, 512, 640, 768, 896, 1024, 1152};
  static const int w1off[8] = {0, 32768, 81920, 147456, 229376, 327680, 442368, 573440};
  static const int w2off[8] = {0, 131072, 425984, 950272, 1769472, 2949120, 4554752, 6651904};

  unsigned short* ws  = (unsigned short*)d_ws;
  unsigned short* we  = ws;                       // 1,048,576 elems
  unsigned short* w1t = ws + (size_t)BATCH * 64;  // 720,896 elems
  unsigned short* w2t = w1t + 720896;             // 9,306,112 elems (~22 MB total)

  // 1) weighted expert vectors -> bf16
  we_kernel<<<(BATCH * 64) / 256, 256, 0, stream>>>(
      (const float*)d_in[0], (const float*)d_in[1], we);

  // 2) convert + transpose weights -> bf16 (B-matrix wants column-contiguous)
  for (int l = 0; l < 8; ++l) {
    const int r = ranks[l], r2 = 2 * r;
    const float* w1 = (const float*)d_in[2 + 4 * l];   // [64 x 2r]
    const float* w2 = (const float*)d_in[4 + 4 * l];   // [2r x r]
    transpose_kernel<<<dim3(r2 / 32, 64 / 32), dim3(32, 8), 0, stream>>>(
        w1, w1t + w1off[l], 64, r2);                   // -> [2r x 64]
    transpose_kernel<<<dim3(r / 32, r2 / 32), dim3(32, 8), 0, stream>>>(
        w2, w2t + w2off[l], r2, r);                    // -> [r x 2r]
  }

  // 3) fused dual-GEMM with LDS-resident h
  Params p;
  p.we = we; p.w1t = w1t; p.w2t = w2t; p.out = (float*)d_out;
  for (int l = 0; l < 8; ++l) {
    p.b1[l] = (const float*)d_in[3 + 4 * l];
    p.b2[l] = (const float*)d_in[5 + 4 * l];
  }
  const int smem = 16 * (2 * 1152 + 8) * 2;  // 73,984 B (<320 KB WGP LDS)
  hipFuncSetAttribute((const void*)fused_kernel,
                      hipFuncAttributeMaxDynamicSharedMemorySize, smem);
  fused_kernel<<<dim3(BATCH / 16, 8), 256, smem, stream>>>(p);
}
